// MemristorLinear_62826781606456
// MI455X (gfx1250) — matrix operand order repaired and checked by reading, not compiled, not run on hardware
//
#include <hip/hip_runtime.h>

// Problem constants (from reference): B=8192, IN=4096, OUT=4096, NVAL=1024
#define B_DIM   8192
#define IN_DIM  4096
#define OUT_DIM 4096
#define NVALS   1024

// GEMM tiling
#define TM 128         // M tile per workgroup
#define TN 128         // N tile per workgroup
#define TK 32          // K per chunk (= one bf16 WMMA K-step)
#define LDS_STRIDE 80  // bytes per LDS tile row: 32 bf16 = 64B data + 16B pad (bank-conflict-free)
#define A_TILE_BYTES (TM * LDS_STRIDE)     // 10240
#define BUF_BYTES    (2 * A_TILE_BYTES)    // A + B tile = 20480
#define KCHUNKS (IN_DIM / TK)              // 128

typedef __attribute__((ext_vector_type(16))) __bf16 v16bf;
typedef __attribute__((ext_vector_type(8)))  float  v8f;

union Frag16 { uint4 q[2]; v16bf v; };

__device__ __forceinline__ unsigned short f2bf(float f) {
  unsigned u = __float_as_uint(f);
  u += 0x7FFFu + ((u >> 16) & 1u);   // round-to-nearest-even
  return (unsigned short)(u >> 16);
}

// ---------------------------------------------------------------------------
// Pass 1a: x fp32 -> bf16 (memory bound, 128MB read / 64MB write)
// ---------------------------------------------------------------------------
__global__ void cvt_x_kernel(const float* __restrict__ x,
                             unsigned short* __restrict__ xb) {
  unsigned i = blockIdx.x * blockDim.x + threadIdx.x;   // i indexes groups of 4
  float4 v = ((const float4*)x)[i];
  ushort4 o;
  o.x = f2bf(v.x); o.y = f2bf(v.y); o.z = f2bf(v.z); o.w = f2bf(v.w);
  ((ushort4*)xb)[i] = o;
}

// ---------------------------------------------------------------------------
// Pass 1b: decode codebook -> dense bf16 W[OUT][IN] (64MB idx read, 32MB write)
// Codebook (4KB) staged in LDS for the gather.
// ---------------------------------------------------------------------------
__global__ void decode_w_kernel(const int* __restrict__ widx,
                                const float* __restrict__ values,
                                unsigned short* __restrict__ wb) {
  __shared__ float vals[NVALS];
  for (int i = threadIdx.x; i < NVALS; i += blockDim.x) vals[i] = values[i];
  __syncthreads();
  unsigned i = blockIdx.x * blockDim.x + threadIdx.x;   // groups of 4
  int4 idx = ((const int4*)widx)[i];
  ushort4 o;
  o.x = f2bf(vals[idx.x & (NVALS - 1)]);
  o.y = f2bf(vals[idx.y & (NVALS - 1)]);
  o.z = f2bf(vals[idx.z & (NVALS - 1)]);
  o.w = f2bf(vals[idx.w & (NVALS - 1)]);
  ((ushort4*)wb)[i] = o;
}

// ---------------------------------------------------------------------------
// Pass 2: out = xb @ wb^T + bias, bf16 WMMA with fp32 accumulation.
// 8 waves: wave (wm in 0..1, wn in 0..3) computes a 64x32 sub-tile as
// 4x2 WMMA 16x16 tiles. Double-buffered LDS fed by async global->LDS loads.
// ---------------------------------------------------------------------------
__global__ __launch_bounds__(256, 2) void gemm_bf16_kernel(
    const unsigned short* __restrict__ xb,
    const unsigned short* __restrict__ wb,
    const float* __restrict__ bias,
    float* __restrict__ out) {
  __shared__ __align__(16) unsigned char smem[2 * BUF_BYTES];   // 40 KB

  const unsigned tid  = threadIdx.x;
  const unsigned lane = tid & 31;
  const unsigned wave = tid >> 5;
  const unsigned wm = wave & 1;    // 0..1  -> M offset wm*64
  const unsigned wn = wave >> 1;   // 0..3  -> N offset wn*32
  const unsigned lrow = lane & 15;
  const unsigned lhi  = lane >> 4;

  const unsigned tile_m = blockIdx.y * TM;
  const unsigned tile_n = blockIdx.x * TN;

  // LDS byte address of smem for the async-load VDST operand
  // (flat LDS-aperture address: low 32 bits == DS offset per ISA aperture rules)
  const unsigned smem_base = (unsigned)(size_t)(const void*)smem;

  // --- loop-carried async-load addresses (advance by TK*2 = 64B per chunk) ---
  // Each thread moves 2x16B of A and 2x16B of B per chunk.
  // Segment tid: row = tid>>2 (of 128), kseg = tid&3; segment tid+256: row += 64.
  const unsigned arow  = tid >> 2;
  const unsigned kseg  = tid & 3u;
  const unsigned ldsA0 = smem_base + arow * LDS_STRIDE + kseg * 16u;
  const unsigned ldsA1 = ldsA0 + 64u * LDS_STRIDE;
  const unsigned ldsB0 = ldsA0 + A_TILE_BYTES;
  const unsigned ldsB1 = ldsA1 + A_TILE_BYTES;

  unsigned long long gA0 =
      (unsigned long long)xb +
      ((unsigned long long)(tile_m + arow) * IN_DIM + kseg * 8u) * 2ull;
  unsigned long long gA1 = gA0 + 64ull * (IN_DIM * 2ull);
  unsigned long long gB0 =
      (unsigned long long)wb +
      ((unsigned long long)(tile_n + arow) * IN_DIM + kseg * 8u) * 2ull;
  unsigned long long gB1 = gB0 + 64ull * (IN_DIM * 2ull);

  auto issue_loads = [&](unsigned bufOff) {
    unsigned a0 = ldsA0 + bufOff, a1 = ldsA1 + bufOff;
    unsigned b0 = ldsB0 + bufOff, b1 = ldsB1 + bufOff;
    asm volatile("global_load_async_to_lds_b128 %0, %1, off" :: "v"(a0), "v"(gA0));
    asm volatile("global_load_async_to_lds_b128 %0, %1, off" :: "v"(a1), "v"(gA1));
    asm volatile("global_load_async_to_lds_b128 %0, %1, off" :: "v"(b0), "v"(gB0));
    asm volatile("global_load_async_to_lds_b128 %0, %1, off" :: "v"(b1), "v"(gB1));
    gA0 += (unsigned long long)(TK * 2);
    gA1 += (unsigned long long)(TK * 2);
    gB0 += (unsigned long long)(TK * 2);
    gB1 += (unsigned long long)(TK * 2);
  };

  // Fragment base byte-offsets within one buffer (ISA 7.12.2 layouts):
  // A (16x32 bf16): lane holds row l%16; K = kb..kb+7 and kb+16..kb+23, kb=(l/16)*8
  const unsigned aoff = (wm * 64u + lrow) * LDS_STRIDE + lhi * 16u;
  // B (32x16 bf16): lane holds col l%16; K = kb..kb+7 and kb+16..kb+23, kb=(l/16)*8
  const unsigned boff = A_TILE_BYTES + (wn * 32u + lrow) * LDS_STRIDE + lhi * 16u;

  v8f acc[4][2] = {};   // 4 M-subtiles x 2 N-subtiles, 8 f32 each

  // Prologue: fill buffer 0
  issue_loads(0u);
  asm volatile("s_wait_asynccnt 0" ::: "memory");
  __syncthreads();

  unsigned curOff = 0u;
#pragma clang loop unroll(disable)
  for (unsigned kc = 0; kc < KCHUNKS; ++kc) {
    const unsigned char* ap = smem + (curOff + aoff);
    const unsigned char* bp = smem + (curOff + boff);

    // Load B fragments first (reused by 4 WMMAs each)
    Frag16 bfrag[2];
#pragma unroll
    for (int ns = 0; ns < 2; ++ns) {
      const uint4* pb = (const uint4*)(bp + ns * (16 * LDS_STRIDE));
      bfrag[ns].q[0] = pb[0];
      bfrag[ns].q[1] = pb[2];
    }
    Frag16 afrag[4];
#pragma unroll
    for (int ms = 0; ms < 4; ++ms) {
      const unsigned char* pa = ap + ms * (16 * LDS_STRIDE);
      afrag[ms].q[0] = *(const uint4*)(pa);
      afrag[ms].q[1] = *(const uint4*)(pa + 32);
    }

    // Overlap: issue next chunk's async loads while ds reads are in flight
    if (kc + 1 < KCHUNKS) issue_loads(curOff ^ BUF_BYTES);

#pragma unroll
    for (int ms = 0; ms < 4; ++ms) {
#pragma unroll
      for (int ns = 0; ns < 2; ++ns) {
        acc[ms][ns] = __builtin_amdgcn_wmma_f32_16x16x32_bf16(
            false, afrag[ms].v, false, bfrag[ns].v,
            (short)0, acc[ms][ns], false, false);
      }
    }

    asm volatile("s_wait_asynccnt 0" ::: "memory");
    __syncthreads();
    curOff ^= BUF_BYTES;
  }

  // Epilogue: C/D layout -> element (m,n): vgpr r = m%8, lane = (m/8)*16 + n
#pragma unroll
  for (int ns = 0; ns < 2; ++ns) {
    const unsigned n = tile_n + wn * 32 + (unsigned)ns * 16 + lrow;
    const float bv = bias[n];
#pragma unroll
    for (int ms = 0; ms < 4; ++ms) {
      const unsigned mbase = tile_m + wm * 64 + (unsigned)ms * 16 + lhi * 8;
#pragma unroll
      for (int r = 0; r < 8; ++r) {
        out[(unsigned long long)(mbase + r) * OUT_DIM + n] = acc[ms][ns][r] + bv;
      }
    }
  }
}

// ---------------------------------------------------------------------------
// Host launch
// ---------------------------------------------------------------------------
extern "C" void kernel_launch(void* const* d_in, const int* in_sizes, int n_in,
                              void* d_out, int out_size, void* d_ws, size_t ws_size,
                              hipStream_t stream) {
  (void)in_sizes; (void)n_in; (void)out_size; (void)ws_size;
  const float* x      = (const float*)d_in[0];
  const float* values = (const float*)d_in[1];
  const float* bias   = (const float*)d_in[2];
  const int*   widx   = (const int*)d_in[3];
  float*       out    = (float*)d_out;

  // Workspace: xb (8192*4096 bf16 = 64MB) | wb (4096*4096 bf16 = 32MB)
  unsigned short* xbuf = (unsigned short*)d_ws;
  unsigned short* wbuf = xbuf + (size_t)B_DIM * IN_DIM;

  {
    const unsigned groups = (unsigned)((size_t)B_DIM * IN_DIM / 4);  // 8,388,608
    cvt_x_kernel<<<groups / 256, 256, 0, stream>>>(x, xbuf);
  }
  {
    const unsigned groups = (unsigned)((size_t)OUT_DIM * IN_DIM / 4); // 4,194,304
    decode_w_kernel<<<groups / 256, 256, 0, stream>>>(widx, values, wbuf);
  }
  {
    dim3 grid(OUT_DIM / TN, B_DIM / TM);   // 32 x 64 = 2048 workgroups
    gemm_bf16_kernel<<<grid, 256, 0, stream>>>(xbuf, wbuf, bias, out);
  }
}
